// DeltaNet_31877247271495
// MI455X (gfx1250) — matrix operand order repaired and checked by reading, not compiled, not run on hardware
//
#include <hip/hip_runtime.h>
#include <hip/hip_bf16.h>

// ---------------------------------------------------------------------------
// Types for WMMA (gfx1250, wave32)
// ---------------------------------------------------------------------------
typedef _Float16 half_t;
typedef __attribute__((ext_vector_type(16))) _Float16 v16h;
typedef __attribute__((ext_vector_type(8)))  _Float16 v8h;
typedef __attribute__((ext_vector_type(8)))  float    v8f;

#define BB 4
#define LL 2048
#define DD 1024
#define HH 4
#define DKK 256
#define DVV 256
#define NTOK (BB * LL)        // 8192
#define NCHUNK 64

__device__ __forceinline__ v8f wmma_f16(v16h a, v16h b, v8f c) {
  return __builtin_amdgcn_wmma_f32_16x16x32_f16(false, a, false, b, (short)0, c,
                                                false, false);
}

// A fragment: 16x32 f16, row-major source with leading dim ld (halves).
// lane<16 holds row m=lane, K = 0..15; lane>=16 holds K = 16..31.
__device__ __forceinline__ v16h frag_a(const half_t* base, int ld, int m0, int k0,
                                       int lane) {
  int m = m0 + (lane & 15);
  int ko = k0 + ((lane >> 4) << 4);
  const half_t* p = base + (size_t)m * ld + ko;
  v8h lo = *(const v8h*)p;
  v8h hi = *(const v8h*)(p + 8);
  v16h f;
#pragma unroll
  for (int i = 0; i < 8; ++i) { f[i] = lo[i]; f[i + 8] = hi[i]; }
  return f;
}

// B fragment: 32x16 f16 (KxN) read from transposed storage Bt[n][k], ld halves.
// lane<16: N=lane, K=0..15 ; lane>=16: N=lane-16, K=16..31 (contiguous 32B).
__device__ __forceinline__ v16h frag_b(const half_t* base, int ld, int n0, int k0,
                                       int lane) {
  int n = n0 + (lane & 15);
  int ko = k0 + ((lane >> 4) << 4);
  return *(const v16h*)(base + (size_t)n * ld + ko);
}

// B fragment gathered from f32 matrix S[k][n] (row-major, ld floats) with cvt.
__device__ __forceinline__ v16h frag_b_f32(const float* S, int ld, int n0, int k0,
                                           int lane) {
  int n = n0 + (lane & 15);
  int ks = k0 + ((lane >> 4) << 4);
  v16h f;
#pragma unroll
  for (int t = 0; t < 16; ++t) f[t] = (_Float16)S[(size_t)(ks + t) * ld + n];
  return f;
}

// ---------------------------------------------------------------------------
// Generic tiled WMMA GEMM:  C[M,N] = act(A[M,K] * B[K,N] + bias)
// AMODE 0: A0 row-major [M,K].
// AMODE 1: gather concat(hs | fir_s | fir_l | delta_out) for the fuse1 GEMM.
// ACT 0: none, 1: exact GELU.
// Block: 256 threads (8 waves), tile 128x128, BK=64, waves as 2(M)x4(N).
// global_prefetch_b8 of the next K-tile overlaps WMMA with HBM fetch.
// ---------------------------------------------------------------------------
template <int AMODE>
__device__ __forceinline__ const float* ptrA4(const float* A0, const float* A1,
                                              const float* A2, const float* A3,
                                              int row, int k, int K) {
  if (AMODE == 0) return A0 + (size_t)row * K + k;
  int src = k >> 10, cc = k & 1023;
  if (src == 0)      return A0 + (size_t)row * 1024 + cc;
  else if (src == 1) return A1 + (size_t)row * 1024 + cc;
  else if (src == 2) return A2 + (size_t)row * 1024 + cc;
  int h = cc >> 8, d = cc & 255;
  int b = row >> 11, l = row & 2047;
  return A3 + (((size_t)(b * HH + h) * LL + l) * DVV + d);
}

#define GA_LD 72
#define GB_LD 80

template <int AMODE, int ACT>
__global__ __launch_bounds__(256) void gemm_wmma_kernel(
    const float* __restrict__ A0, const float* __restrict__ A1,
    const float* __restrict__ A2, const float* __restrict__ A3,
    const float* __restrict__ B, const float* __restrict__ bias,
    float* __restrict__ C, int M, int N, int K) {
  __shared__ __attribute__((aligned(32))) half_t Ah[128 * GA_LD];
  __shared__ __attribute__((aligned(32))) half_t Bt[128 * GB_LD];

  int tid = threadIdx.x, lane = tid & 31, wv = tid >> 5;
  int m0 = blockIdx.y * 128, n0 = blockIdx.x * 128;
  int wm = (wv & 1) * 64, wn = (wv >> 1) * 32;

  v8f acc[4][2];
#pragma unroll
  for (int i = 0; i < 4; ++i)
#pragma unroll
    for (int j = 0; j < 2; ++j) acc[i][j] = (v8f)0.f;

  for (int k0 = 0; k0 < K; k0 += 64) {
    // stage A tile 128x64 (f32 -> f16), row-major ld 72
#pragma unroll
    for (int t = 0; t < 8; ++t) {
      int idx = t * 256 + tid;
      int r = idx >> 4, kq = (idx & 15) * 4;
      float4 fv = *(const float4*)ptrA4<AMODE>(A0, A1, A2, A3, m0 + r, k0 + kq, K);
      Ah[r * GA_LD + kq + 0] = (_Float16)fv.x;
      Ah[r * GA_LD + kq + 1] = (_Float16)fv.y;
      Ah[r * GA_LD + kq + 2] = (_Float16)fv.z;
      Ah[r * GA_LD + kq + 3] = (_Float16)fv.w;
    }
    // stage B tile 64x128 transposed -> Bt[n][k], ld 80
#pragma unroll
    for (int t = 0; t < 8; ++t) {
      int idx = t * 256 + tid;
      int kk = idx >> 5, nq = (idx & 31) * 4;
      float4 fv = *(const float4*)(B + (size_t)(k0 + kk) * N + n0 + nq);
      Bt[(nq + 0) * GB_LD + kk] = (_Float16)fv.x;
      Bt[(nq + 1) * GB_LD + kk] = (_Float16)fv.y;
      Bt[(nq + 2) * GB_LD + kk] = (_Float16)fv.z;
      Bt[(nq + 3) * GB_LD + kk] = (_Float16)fv.w;
    }
    // prefetch next K tile into caches (global_prefetch_b8)
    if (k0 + 64 < K) {
      int pr = tid >> 1, pko = (tid & 1) * 32;
      __builtin_prefetch(ptrA4<AMODE>(A0, A1, A2, A3, m0 + pr, k0 + 64 + pko, K), 0, 3);
      if (tid < 128) {
        int pkk = tid >> 1, pnq = (tid & 1) * 64;
        __builtin_prefetch(B + (size_t)(k0 + 64 + pkk) * N + n0 + pnq, 0, 3);
      }
    }
    __syncthreads();

#pragma unroll
    for (int ks = 0; ks < 64; ks += 32) {
      v16h bfr[2];
#pragma unroll
      for (int j = 0; j < 2; ++j) bfr[j] = frag_b(Bt, GB_LD, wn + 16 * j, ks, lane);
#pragma unroll
      for (int i = 0; i < 4; ++i) {
        v16h afr = frag_a(Ah, GA_LD, wm + 16 * i, ks, lane);
#pragma unroll
        for (int j = 0; j < 2; ++j) acc[i][j] = wmma_f16(afr, bfr[j], acc[i][j]);
      }
    }
    __syncthreads();
  }

#pragma unroll
  for (int i = 0; i < 4; ++i)
#pragma unroll
    for (int j = 0; j < 2; ++j)
#pragma unroll
      for (int e = 0; e < 8; ++e) {
        int gr = m0 + wm + 16 * i + e + ((lane >> 4) << 3);
        int gc = n0 + wn + 16 * j + (lane & 15);
        float v = acc[i][j][e];
        if (bias) v += bias[gc];
        if (ACT == 1) v = 0.5f * v * (1.f + erff(v * 0.70710678118654752f));
        C[(size_t)gr * N + gc] = v;
      }
}

// ---------------------------------------------------------------------------
// Depthwise causal conv (taps in {3,4,31}) with optional SiLU
// ---------------------------------------------------------------------------
__global__ __launch_bounds__(256) void dwconv_kernel(
    const float* __restrict__ x, const float* __restrict__ w,
    float* __restrict__ y, int L, int C, int taps, int do_silu) {
  size_t i = (size_t)blockIdx.x * 256 + threadIdx.x;
  int c = (int)(i % C);
  size_t bl = i / C;
  int l = (int)(bl % L);
  size_t base = (bl - l) * C;
  float acc = 0.f;
  for (int j = 0; j < taps; ++j) {
    int li = l - (taps - 1) + j;
    if (li >= 0) acc += x[base + (size_t)li * C + c] * w[c * taps + j];
  }
  if (do_silu) acc = acc / (1.f + __expf(-acc));
  y[i] = acc;
}

// ---------------------------------------------------------------------------
// beta = sigmoid(hs @ W_b)   (one wave per token)
// ---------------------------------------------------------------------------
__global__ __launch_bounds__(256) void beta_kernel(const float* __restrict__ hs,
                                                   const float* __restrict__ Wb,
                                                   float* __restrict__ beta) {
  int tid = threadIdx.x, lane = tid & 31, wv = tid >> 5;
  size_t t = (size_t)blockIdx.x * 8 + wv;
  const float* hrow = hs + t * DD;
  for (int h = 0; h < HH; ++h) {
    float acc = 0.f;
    for (int k = lane; k < DD; k += 32) acc += hrow[k] * Wb[k * HH + h];
#pragma unroll
    for (int o = 16; o > 0; o >>= 1) acc += __shfl_xor(acc, o, 32);
    if (lane == 0) beta[t * HH + h] = 1.f / (1.f + __expf(-acc));
  }
}

// ---------------------------------------------------------------------------
// Delta-rule per-chunk preprocessing: one workgroup per (b,h,chunk).
// l2norm(q), l2norm(k), kb = k*beta, v *= beta,
// att = UT-transform(-(kb k^T) strict-lower) + I, u = att@v, w = att@kb.
// Outputs qn,kn,u,w as f16 in [b,h,l,256] layout.
// ---------------------------------------------------------------------------
#define PRE_KN_LD 272
#define PRE_KB_LD 264
constexpr size_t PRE_LDS =
    (size_t)(32 * PRE_KN_LD + 32 * PRE_KB_LD + 256 * 48 + 256 * 48 + 32 * 40) * 2 +
    (size_t)32 * 33 * 4 + 32 * 4;

__global__ __launch_bounds__(256) void delta_pre_kernel(
    const float* __restrict__ qc, const float* __restrict__ kc,
    const float* __restrict__ vc, const float* __restrict__ betag,
    half_t* __restrict__ qn_g, half_t* __restrict__ kn_g,
    half_t* __restrict__ u_g, half_t* __restrict__ w_g) {
  extern __shared__ char smem[];
  half_t* kn   = (half_t*)smem;                 // 32 x 272
  half_t* kb   = kn + 32 * PRE_KN_LD;           // 32 x 264
  half_t* vT   = kb + 32 * PRE_KB_LD;           // 256 x 48 (transposed)
  half_t* kbT  = vT + 256 * 48;                 // 256 x 48 (transposed)
  half_t* atth = kbT + 256 * 48;                // 32 x 40
  float*  att  = (float*)(atth + 32 * 40);      // 32 x 33
  float*  betas = att + 32 * 33;                // 32

  int bc = blockIdx.x;
  int n = bc & 63, h = (bc >> 6) & 3, b = bc >> 8;
  int tid = threadIdx.x, lane = tid & 31, wv = tid >> 5;
  size_t tok0 = (size_t)b * LL + n * 32;
  const float* qrow = qc + tok0 * DD + h * DKK;
  const float* krow = kc + tok0 * DD + h * DKK;
  const float* vrow = vc + tok0 * DD + h * DVV;
  size_t obase = ((size_t)(b * HH + h) * LL + n * 32) * DKK;

  if (tid < 32) betas[tid] = betag[(tok0 + tid) * HH + h];
  __syncthreads();

  for (int it = 0; it < 4; ++it) {
    int r = wv + 8 * it;
    float bv = betas[r];
    // ---- k: l2norm, kn, kb, kbT ----
    {
      float x[8], ss = 0.f;
#pragma unroll
      for (int e = 0; e < 8; ++e) {
        x[e] = krow[(size_t)r * DD + lane + 32 * e];
        ss += x[e] * x[e];
      }
#pragma unroll
      for (int o = 16; o > 0; o >>= 1) ss += __shfl_xor(ss, o, 32);
      float rs = rsqrtf(ss + 1e-6f);
#pragma unroll
      for (int e = 0; e < 8; ++e) {
        int d = lane + 32 * e;
        float knv = x[e] * rs, kbv = knv * bv;
        kn[r * PRE_KN_LD + d] = (_Float16)knv;
        kb[r * PRE_KB_LD + d] = (_Float16)kbv;
        kbT[d * 48 + r] = (_Float16)kbv;
        kn_g[obase + (size_t)r * DKK + d] = (_Float16)knv;
      }
    }
    // ---- q: l2norm -> global ----
    {
      float x[8], ss = 0.f;
#pragma unroll
      for (int e = 0; e < 8; ++e) {
        x[e] = qrow[(size_t)r * DD + lane + 32 * e];
        ss += x[e] * x[e];
      }
#pragma unroll
      for (int o = 16; o > 0; o >>= 1) ss += __shfl_xor(ss, o, 32);
      float rs = rsqrtf(ss + 1e-6f);
#pragma unroll
      for (int e = 0; e < 8; ++e)
        qn_g[obase + (size_t)r * DKK + lane + 32 * e] = (_Float16)(x[e] * rs);
    }
    // ---- v: scale by beta, store transposed ----
#pragma unroll
    for (int e = 0; e < 8; ++e) {
      int d = lane + 32 * e;
      vT[d * 48 + r] = (_Float16)(vrow[(size_t)r * DD + d] * bv);
    }
  }
  __syncthreads();

  // att = -(kb @ kn^T), strict lower only
  if (wv < 4) {
    int mt = (wv & 1) * 16, nt = (wv >> 1) * 16;
    v8f acc = (v8f)0.f;
    for (int kk = 0; kk < 256; kk += 32) {
      v16h a = frag_a(kb, PRE_KB_LD, mt, kk, lane);
      v16h bf = frag_b(kn, PRE_KN_LD, nt, kk, lane);
      acc = wmma_f16(a, bf, acc);
    }
#pragma unroll
    for (int e = 0; e < 8; ++e) {
      int i = mt + e + ((lane >> 4) << 3);
      int j = nt + (lane & 15);
      att[i * 33 + j] = (j < i) ? -acc[e] : 0.f;
    }
  }
  __syncthreads();

  // UT transform: forward substitution (wave 0, sequential in i)
  if (wv == 0) {
    for (int i = 1; i < 32; ++i) {
      float up = 0.f;
      for (int t = 0; t < i; ++t) up += att[i * 33 + t] * att[t * 33 + lane];
      if (lane < i) att[i * 33 + lane] += up;
    }
  }
  __syncthreads();

  // att += I, convert to f16
  for (int idx = tid; idx < 1024; idx += 256) {
    int i = idx >> 5, j = idx & 31;
    atth[i * 40 + j] = (_Float16)(att[i * 33 + j] + (i == j ? 1.f : 0.f));
  }
  __syncthreads();

  // u = att @ v  (vT), w = att @ kb (kbT) : 64 tile jobs over 8 waves
  for (int jb = wv; jb < 64; jb += 8) {
    int which = jb >> 5;
    int r = jb & 31;
    int mt = (r >> 4) * 16, nt = (r & 15) * 16;
    const half_t* Bm = which ? kbT : vT;
    v16h a = frag_a(atth, 40, mt, 0, lane);
    v16h bf = frag_b(Bm, 48, nt, 0, lane);
    v8f acc = (v8f)0.f;
    acc = wmma_f16(a, bf, acc);
    half_t* Og = which ? w_g : u_g;
#pragma unroll
    for (int e = 0; e < 8; ++e) {
      int m = mt + e + ((lane >> 4) << 3);
      int nn = nt + (lane & 15);
      Og[obase + (size_t)m * DKK + nn] = (_Float16)acc[e];
    }
  }
}

// ---------------------------------------------------------------------------
// Delta-rule sequential scan: one workgroup per (b,h), S[256][256] f32 in LDS.
// Per chunk: u_adj = u - w@S ; local = lower(q k^T) ; o = q@S + local@u_adj ;
//            S += k^T @ u_adj.   Next chunk operands prefetched during compute.
// ---------------------------------------------------------------------------
#define S_LD 260
constexpr size_t SCAN_LDS = (size_t)256 * S_LD * 4 +
                            (size_t)(256 * 48 + 256 * 48 + 32 * 48) * 2;

__global__ __launch_bounds__(512) void delta_scan_kernel(
    const half_t* __restrict__ qn_g, const half_t* __restrict__ kn_g,
    const half_t* __restrict__ u_g, const half_t* __restrict__ w_g,
    float* __restrict__ o_g) {
  extern __shared__ char smem[];
  float*  S      = (float*)smem;             // 256 x 260
  half_t* kT     = (half_t*)(S + 256 * S_LD);// 256 x 48 : k_i^T
  half_t* uT     = kT + 256 * 48;            // 256 x 48 : u_adj^T
  half_t* localh = uT + 256 * 48;            // 32 x 48

  int tid = threadIdx.x, lane = tid & 31, wv = tid >> 5;
  size_t bh = blockIdx.x;

  for (int i = tid; i < 256 * S_LD; i += 512) S[i] = 0.f;
  __syncthreads();

  for (int n = 0; n < NCHUNK; ++n) {
    size_t base = (bh * LL + (size_t)n * 32) * DKK;
    const half_t* qg = qn_g + base;
    const half_t* kg = kn_g + base;
    const half_t* ug = u_g + base;
    const half_t* wg = w_g + base;
    float* og = o_g + base;

    // prefetch next chunk's operands while this chunk computes
    if (n + 1 < NCHUNK) {
      size_t nb = base + 32 * DKK + (size_t)tid * 16;
      __builtin_prefetch(qn_g + nb, 0, 3);
      __builtin_prefetch(kn_g + nb, 0, 3);
      __builtin_prefetch(u_g + nb, 0, 3);
      __builtin_prefetch(w_g + nb, 0, 3);
    }

    // build k_i^T
    for (int i = tid; i < 32 * 256; i += 512) {
      int d = i & 255, r = i >> 8;
      kT[d * 48 + r] = kg[(size_t)r * 256 + d];
    }
    __syncthreads();

    // phase 1: jobs 0..31 -> u_adj tiles (w@S), 32..35 -> local tiles (q k^T)
    for (int jb = wv; jb < 36; jb += 16) {
      if (jb < 32) {
        int mt = (jb & 1) * 16, nt = (jb >> 1) * 16;
        v8f acc = (v8f)0.f;
        for (int kk = 0; kk < 256; kk += 32) {
          v16h a = frag_a(wg, 256, mt, kk, lane);
          v16h bf = frag_b_f32(S, S_LD, nt, kk, lane);
          acc = wmma_f16(a, bf, acc);
        }
#pragma unroll
        for (int e = 0; e < 8; ++e) {
          int m = mt + e + ((lane >> 4) << 3);
          int nn = nt + (lane & 15);
          float ua = (float)ug[(size_t)m * 256 + nn] - acc[e];
          uT[nn * 48 + m] = (_Float16)ua;
        }
      } else {
        int t2 = jb - 32;
        int mt = (t2 & 1) * 16, nt = (t2 >> 1) * 16;
        v8f acc = (v8f)0.f;
        for (int kk = 0; kk < 256; kk += 32) {
          v16h a = frag_a(qg, 256, mt, kk, lane);
          v16h bf = frag_b(kg, 256, nt, kk, lane);
          acc = wmma_f16(a, bf, acc);
        }
#pragma unroll
        for (int e = 0; e < 8; ++e) {
          int i = mt + e + ((lane >> 4) << 3);
          int j = nt + (lane & 15);
          localh[i * 48 + j] = (_Float16)((j <= i) ? acc[e] : 0.f);
        }
      }
    }
    __syncthreads();

    // phase 2: o = q@S + local@u_adj
    for (int jb = wv; jb < 32; jb += 16) {
      int mt = (jb & 1) * 16, nt = (jb >> 1) * 16;
      v8f acc = (v8f)0.f;
      for (int kk = 0; kk < 256; kk += 32) {
        v16h a = frag_a(qg, 256, mt, kk, lane);
        v16h bf = frag_b_f32(S, S_LD, nt, kk, lane);
        acc = wmma_f16(a, bf, acc);
      }
      {
        v16h a = frag_a(localh, 48, mt, 0, lane);
        v16h bf = frag_b(uT, 48, nt, 0, lane);
        acc = wmma_f16(a, bf, acc);
      }
#pragma unroll
      for (int e = 0; e < 8; ++e) {
        int m = mt + e + ((lane >> 4) << 3);
        int nn = nt + (lane & 15);
        og[(size_t)m * 256 + nn] = acc[e];
      }
    }
    __syncthreads();

    // phase 3: S += k^T @ u_adj  (16 dk-tiles x 16 dv-tiles; one row per wave)
    {
      int mt = wv * 16;
      for (int nt = 0; nt < 256; nt += 16) {
        v8f c;
#pragma unroll
        for (int e = 0; e < 8; ++e)
          c[e] = S[(size_t)(mt + e + ((lane >> 4) << 3)) * S_LD + nt + (lane & 15)];
        v16h a = frag_a(kT, 48, mt, 0, lane);
        v16h bf = frag_b(uT, 48, nt, 0, lane);
        c = wmma_f16(a, bf, c);
#pragma unroll
        for (int e = 0; e < 8; ++e)
          S[(size_t)(mt + e + ((lane >> 4) << 3)) * S_LD + nt + (lane & 15)] = c[e];
      }
    }
    __syncthreads();
  }
}

// ---------------------------------------------------------------------------
// fuse2: logits = hid @ W_fuse2 + b  (N=16, one wave per token)
// ---------------------------------------------------------------------------
__global__ __launch_bounds__(256) void fuse2_kernel(
    const float* __restrict__ hid, const float* __restrict__ W2,
    const float* __restrict__ b2, float* __restrict__ logits) {
  int tid = threadIdx.x, lane = tid & 31, wv = tid >> 5;
  size_t t = (size_t)blockIdx.x * 8 + wv;
  int j = lane & 15, ks = lane >> 4;
  const float* hrow = hid + t * 2048;
  float acc = 0.f;
  for (int k = ks; k < 2048; k += 2) acc += hrow[k] * W2[k * 16 + j];
  acc += __shfl_xor(acc, 16, 32);
  if (lane < 16) logits[t * 16 + j] = acc + b2[j];
}

// ---------------------------------------------------------------------------
// Gate mixing + RMSNorm.  One block per token; 64 threads per head.
// ---------------------------------------------------------------------------
__global__ __launch_bounds__(256) void mix_norm_kernel(
    const float* __restrict__ logits, const float* __restrict__ log_tau,
    const float* __restrict__ vc, const float* __restrict__ fir_s,
    const float* __restrict__ fir_l, const float* __restrict__ dout,
    const float* __restrict__ norm_w, float* __restrict__ o_n) {
  __shared__ float red[256];
  int t = blockIdx.x;
  int b = t >> 11, l = t & 2047;
  int tid = threadIdx.x;
  int h = tid >> 6, q = tid & 63;
  int d0 = q * 4;

  const float* lg = logits + (size_t)t * 16 + h * 4;
  float tau = __expf(log_tau[h]);
  float p_raw = 1.f / (1.f + __expf(-lg[3]));
  float fl = 0.01f + 0.09f * (1.f - p_raw);
  float p_val = (1.f - fl) * p_raw;
  float x0 = lg[0] / tau, x1 = lg[1] / tau, x2 = lg[2] / tau;
  float mx = fmaxf(x0, fmaxf(x1, x2));
  float e0 = __expf(x0 - mx), e1 = __expf(x1 - mx), e2 = __expf(x2 - mx);
  float inv = (1.f - p_val) / (e0 + e1 + e2);
  float p0 = e0 * inv, p1 = e1 * inv, p2 = e2 * inv;

  size_t vb = (size_t)t * 1024 + h * 256 + d0;
  size_t db = ((size_t)(b * HH + h) * LL + l) * 256 + d0;
  float o[4], ss = 0.f;
#pragma unroll
  for (int j = 0; j < 4; ++j) {
    float ov = p0 * fir_s[vb + j] + p1 * fir_l[vb + j] + p2 * dout[db + j] +
               p_val * vc[vb + j];
    o[j] = ov;
    ss += ov * ov;
  }
  red[tid] = ss;
  __syncthreads();
  for (int ofs = 32; ofs > 0; ofs >>= 1) {
    if (q < ofs) red[tid] += red[tid + ofs];
    __syncthreads();
  }
  float scale = rsqrtf(red[h * 64] / 256.f + 1e-5f);
#pragma unroll
  for (int j = 0; j < 4; ++j) o_n[vb + j] = o[j] * scale * norm_w[d0 + j];
}

// ---------------------------------------------------------------------------
// Host-side launch
// ---------------------------------------------------------------------------
extern "C" void kernel_launch(void* const* d_in, const int* in_sizes, int n_in,
                              void* d_out, int out_size, void* d_ws,
                              size_t ws_size, hipStream_t stream) {
  (void)in_sizes; (void)n_in; (void)out_size; (void)ws_size;
  const float* hs       = (const float*)d_in[0];
  const float* W_q      = (const float*)d_in[1];
  const float* W_k      = (const float*)d_in[2];
  const float* W_v      = (const float*)d_in[3];
  const float* W_b      = (const float*)d_in[4];
  const float* cw_q     = (const float*)d_in[5];
  const float* cw_k     = (const float*)d_in[6];
  const float* cw_v     = (const float*)d_in[7];
  const float* fir_sw   = (const float*)d_in[8];
  const float* fir_lw   = (const float*)d_in[9];
  const float* W_f1     = (const float*)d_in[10];
  const float* b_f1     = (const float*)d_in[11];
  const float* W_f2     = (const float*)d_in[12];
  const float* b_f2     = (const float*)d_in[13];
  const float* log_tau  = (const float*)d_in[14];
  const float* norm_w   = (const float*)d_in[15];
  const float* W_o      = (const float*)d_in[16];
  float* out = (float*)d_out;

  char* ws = (char*)d_ws;
  constexpr size_t SZ_ACT  = (size_t)NTOK * 1024 * 4;  // 33.5 MB
  constexpr size_t SZ_HALF = (size_t)NTOK * 1024 * 2;
  size_t off = 0;
  auto take = [&](size_t bytes) {
    size_t o = off;
    off = (off + bytes + 255) & ~(size_t)255;
    return o;
  };
  size_t o_qp = take(SZ_ACT), o_kp = take(SZ_ACT), o_vp = take(SZ_ACT);
  size_t o_qc = take(SZ_ACT), o_kc = take(SZ_ACT), o_vc = take(SZ_ACT);
  size_t o_beta = take((size_t)NTOK * HH * 4);
  size_t o_qn = take(SZ_HALF), o_kn = take(SZ_HALF);
  size_t o_u = take(SZ_HALF), o_w = take(SZ_HALF);
  size_t o_do = take(SZ_ACT);
  size_t o_hid = take((size_t)NTOK * 2048 * 4);
  size_t o_lg = take((size_t)NTOK * 16 * 4);
  // reuse raw-projection buffers after convs consume them
  size_t o_fs = o_qp, o_flg = o_kp, o_on = o_vp;

  float* qp = (float*)(ws + o_qp);   float* kp = (float*)(ws + o_kp);
  float* vp = (float*)(ws + o_vp);
  float* qc = (float*)(ws + o_qc);   float* kc = (float*)(ws + o_kc);
  float* vc = (float*)(ws + o_vc);
  float* betag = (float*)(ws + o_beta);
  half_t* qn = (half_t*)(ws + o_qn); half_t* kn = (half_t*)(ws + o_kn);
  half_t* ug = (half_t*)(ws + o_u);  half_t* wg = (half_t*)(ws + o_w);
  float* dov = (float*)(ws + o_do);
  float* hid = (float*)(ws + o_hid);
  float* lgits = (float*)(ws + o_lg);
  float* firs = (float*)(ws + o_fs); float* firl = (float*)(ws + o_flg);
  float* on = (float*)(ws + o_on);

  (void)hipFuncSetAttribute((const void*)delta_pre_kernel,
                            hipFuncAttributeMaxDynamicSharedMemorySize,
                            (int)PRE_LDS);
  (void)hipFuncSetAttribute((const void*)delta_scan_kernel,
                            hipFuncAttributeMaxDynamicSharedMemorySize,
                            (int)SCAN_LDS);

  dim3 gP(1024 / 128, NTOK / 128);  // N=1024 GEMMs
  // 1) q/k/v projections
  gemm_wmma_kernel<0, 0><<<gP, 256, 0, stream>>>(hs, nullptr, nullptr, nullptr,
                                                 W_q, nullptr, qp, NTOK, 1024, 1024);
  gemm_wmma_kernel<0, 0><<<gP, 256, 0, stream>>>(hs, nullptr, nullptr, nullptr,
                                                 W_k, nullptr, kp, NTOK, 1024, 1024);
  gemm_wmma_kernel<0, 0><<<gP, 256, 0, stream>>>(hs, nullptr, nullptr, nullptr,
                                                 W_v, nullptr, vp, NTOK, 1024, 1024);
  // 2) causal dwconv(k=4) + SiLU
  int convBlocks = (int)((size_t)NTOK * 1024 / 256);
  dwconv_kernel<<<convBlocks, 256, 0, stream>>>(qp, cw_q, qc, LL, 1024, 4, 1);
  dwconv_kernel<<<convBlocks, 256, 0, stream>>>(kp, cw_k, kc, LL, 1024, 4, 1);
  dwconv_kernel<<<convBlocks, 256, 0, stream>>>(vp, cw_v, vc, LL, 1024, 4, 1);
  // 3) FIR paths on v (write over raw q/k projections)
  dwconv_kernel<<<convBlocks, 256, 0, stream>>>(vc, fir_sw, firs, LL, 1024, 3, 0);
  dwconv_kernel<<<convBlocks, 256, 0, stream>>>(vc, fir_lw, firl, LL, 1024, 31, 0);
  // 4) beta
  beta_kernel<<<NTOK / 8, 256, 0, stream>>>(hs, W_b, betag);
  // 5) delta-rule chunk preprocessing (1024 chunks)
  delta_pre_kernel<<<BB * HH * NCHUNK, 256, (int)PRE_LDS, stream>>>(
      qc, kc, vc, betag, qn, kn, ug, wg);
  // 6) sequential scan with S resident in LDS (16 workgroups)
  delta_scan_kernel<<<BB * HH, 512, (int)SCAN_LDS, stream>>>(qn, kn, ug, wg, dov);
  // 7) fused gate MLP layer 1 (gathered concat A), GELU
  dim3 gF1(2048 / 128, NTOK / 128);
  gemm_wmma_kernel<1, 1><<<gF1, 256, 0, stream>>>(hs, firs, firl, dov, W_f1,
                                                  b_f1, hid, NTOK, 2048, 4096);
  // 8) gate MLP layer 2 (N=16)
  fuse2_kernel<<<NTOK / 8, 256, 0, stream>>>(hid, W_f2, b_f2, lgits);
  // 9) mixing + RMSNorm
  mix_norm_kernel<<<NTOK, 256, 0, stream>>>(lgits, log_tau, vc, firs, firl, dov,
                                            norm_w, on);
  // 10) output projection -> d_out
  gemm_wmma_kernel<0, 0><<<gP, 256, 0, stream>>>(on, nullptr, nullptr, nullptr,
                                                 W_o, nullptr, out, NTOK, 1024, 1024);
}
